// Linear_68375879352329
// MI455X (gfx1250) — matrix operand order repaired and checked by reading, not compiled, not run on hardware
//
#include <hip/hip_runtime.h>
#include <hip/hip_bf16.h>
#include <cstdint>

// MoE-LoRA fused layer for MI455X (gfx1250, wave32, WMMA + async-to-LDS).
// out = x @ base_W + SCALING * delta  ==  [x | u] @ [[base_W],[B_flat]]
// with u[n, (e,h,r)] = SCALING * gate[n,e] * head_w[n,e,h] * a[n,e,r].
// Main GEMM: [N, 4480] x [4480, 4096] f16, f32 WMMA accumulation,
// 128x256 block, double-buffered LDS filled by GLOBAL_LOAD_ASYNC_TO_LDS_B128.

typedef __attribute__((ext_vector_type(16))) _Float16 v16h;
typedef __attribute__((ext_vector_type(8)))  _Float16 v8h;
typedef __attribute__((ext_vector_type(4)))  _Float16 h4;
typedef __attribute__((ext_vector_type(8)))  float    v8f;
typedef __attribute__((ext_vector_type(4)))  int      v4i;

#define D_IN 4096
#define O_OUT 4096
#define K_TOT 4480      // 4096 + E*H*R (8*3*16=384)
#define N_E 8
#define N_H 3
#define N_R 16
#define N_J 384         // E*H*R
#define N_S 160         // 8 router + 24 head + 128 a
#define SCALING 2.0f

#define BM 128
#define BN 256
#define LDST 40         // LDS row stride in halves (32 + 8 pad, keeps 16B align)
#define A_ROWS 128
#define B_ROWS 256
#define BUF_HALVES ((A_ROWS + B_ROWS) * LDST)   // 15360 halves = 30720 B

// ---- CDNA5 async global->LDS copy (ASYNCcnt) --------------------------------
typedef __attribute__((address_space(1))) v4i* gptr_v4i;
typedef __attribute__((address_space(3))) v4i* lptr_v4i;

__device__ __forceinline__ void async_b128(const _Float16* g, uint32_t lds_off) {
#if __has_builtin(__builtin_amdgcn_global_load_async_to_lds_b128)
  __builtin_amdgcn_global_load_async_to_lds_b128(
      (gptr_v4i)(uintptr_t)g, (lptr_v4i)(uintptr_t)lds_off, 0, 0);
#else
  unsigned long long ga = (unsigned long long)(uintptr_t)g;
  asm volatile("global_load_async_to_lds_b128 %0, %1, off"
               :: "v"(lds_off), "v"(ga) : "memory");
#endif
}

template <int N>
__device__ __forceinline__ void wait_asynccnt() {
#if __has_builtin(__builtin_amdgcn_s_wait_asynccnt)
  __builtin_amdgcn_s_wait_asynccnt(N);
#else
  asm volatile("s_wait_asynccnt %0" :: "i"(N) : "memory");
#endif
}

// ---------------------------------------------------------------- convert x
__global__ __launch_bounds__(256) void convert_x(const float* __restrict__ x,
                                                 _Float16* __restrict__ xcat, int N) {
  int idx = blockIdx.x * 256 + threadIdx.x;          // 4 elements per thread
  int total = N * (D_IN / 4);
  if (idx >= total) return;
  int n  = idx >> 10;                                // / (4096/4)
  int d4 = (idx & 1023) << 2;
  float4 v = *(const float4*)(x + (size_t)n * D_IN + d4);
  h4 o = { (_Float16)v.x, (_Float16)v.y, (_Float16)v.z, (_Float16)v.w };
  *(h4*)(xcat + (size_t)n * K_TOT + d4) = o;
}

// --------------------------------------------- tiled transpose + f32->f16
__global__ __launch_bounds__(256) void transpose_cvt(const float* __restrict__ in,
                                                     _Float16* __restrict__ out,
                                                     int in_cols, int out_ld) {
  __shared__ float tile[32][33];
  int tx = threadIdx.x & 31;
  int ty = threadIdx.x >> 5;
  int c = blockIdx.x * 32 + tx;
#pragma unroll
  for (int i = 0; i < 4; ++i) {
    int r = blockIdx.y * 32 + ty + i * 8;
    tile[ty + i * 8][tx] = in[(size_t)r * in_cols + c];
  }
  __syncthreads();
#pragma unroll
  for (int i = 0; i < 4; ++i) {
    int orow = blockIdx.x * 32 + ty + i * 8;
    out[(size_t)orow * out_ld + blockIdx.y * 32 + tx] = (_Float16)tile[tx][ty + i * 8];
  }
}

// ------------------------------------------- pack router_W / R / A -> f16
__global__ __launch_bounds__(256) void build_mcat(const float* __restrict__ RW,
                                                  const float* __restrict__ Rr,
                                                  const float* __restrict__ A,
                                                  _Float16* __restrict__ Mt) {
  int idx = blockIdx.x * 256 + threadIdx.x;
  if (idx >= N_S * D_IN) return;
  int j = idx >> 12;
  int d = idx & 4095;
  float v;
  if (j < 8) {
    v = RW[(size_t)j * D_IN + d];
  } else if (j < 32) {
    int jj = j - 8; int e = jj / 3; int h = jj - 3 * e;
    v = Rr[((size_t)e * D_IN + d) * N_H + h];
  } else {
    int jj = j - 32; int e = jj >> 4; int r = jj & 15;
    v = A[((size_t)e * D_IN + d) * N_R + r];
  }
  Mt[idx] = (_Float16)v;
}

// -------------------------------------------------- prep GEMM  S = x @ Mt^T
__global__ __launch_bounds__(256) void gemm_prep(const _Float16* __restrict__ X,
                                                 const _Float16* __restrict__ Mt,
                                                 float* __restrict__ S, int N) {
  int wave = threadIdx.x >> 5;
  int lane = threadIdx.x & 31;
  int tile = blockIdx.x * 8 + wave;
  int tm = tile / (N_S / 16), tn = tile % (N_S / 16);
  if (tm * 16 >= N) return;                          // uniform per wave
  int ln = lane & 15, g = lane >> 4;
  const _Float16* arow = X  + (size_t)(tm * 16 + ln) * K_TOT;
  const _Float16* brow = Mt + (size_t)(tn * 16 + ln) * D_IN;
  v8f acc = {};
  for (int k = 0; k < D_IN; k += 32) {
    v16h a, b;
    *(v8h*)&a       = *(const v8h*)(arow + k + g * 8);
    *((v8h*)&a + 1) = *(const v8h*)(arow + k + 16 + g * 8);
    *(v8h*)&b       = *(const v8h*)(brow + k + g * 8);
    *((v8h*)&b + 1) = *(const v8h*)(brow + k + 16 + g * 8);
    acc = __builtin_amdgcn_wmma_f32_16x16x32_f16(false, a, false, b, (short)0, acc, false, false);
  }
  int col = tn * 16 + ln;
  int row0 = tm * 16 + g * 8;
#pragma unroll
  for (int r = 0; r < 8; ++r) S[(size_t)(row0 + r) * N_S + col] = acc[r];
}

// --------------------------------- gating + combine: write u into xcat tail
__global__ __launch_bounds__(256) void gate_combine(const float* __restrict__ S,
                                                    _Float16* __restrict__ xcat, int N) {
  int n = blockIdx.x * 8 + (threadIdx.x >> 5);
  if (n >= N) return;
  int lane = threadIdx.x & 31;
  const float* s = S + (size_t)n * N_S;
  float l1 = -1e30f, l2 = -1e30f; int i1 = -1, i2 = -1;
#pragma unroll
  for (int e = 0; e < N_E; ++e) {
    float v = s[e];
    if (v > l1) { l2 = l1; i2 = i1; l1 = v; i1 = e; }
    else if (v > l2) { l2 = v; i2 = e; }
  }
  float g1 = 1.0f / (1.0f + __expf(l2 - l1));
  float g2 = 1.0f - g1;
  _Float16* u = xcat + (size_t)n * K_TOT + D_IN;
#pragma unroll
  for (int k = 0; k < N_J / 32; ++k) {
    int j = lane + 32 * k;
    int e = j / 48; int rem = j - 48 * e; int h = rem >> 4; int r = rem & 15;
    float gate = (e == i1) ? g1 : ((e == i2) ? g2 : 0.0f);
    float h0 = s[8 + e * 3 + 0], h1 = s[8 + e * 3 + 1], h2 = s[8 + e * 3 + 2];
    float mh = fmaxf(h0, fmaxf(h1, h2));
    float e0 = __expf(h0 - mh), e1 = __expf(h1 - mh), e2 = __expf(h2 - mh);
    float hw = ((h == 0) ? e0 : (h == 1) ? e1 : e2) / (e0 + e1 + e2);
    float a = s[32 + e * 16 + r];
    u[j] = (_Float16)(SCALING * gate * hw * a);
  }
}

// ------------------------------------------------------- main fused GEMM
// 128x256 block, 8 waves (2x4), each wave 64x64 = 4x4 WMMA tiles, BK=32.
// Double-buffered LDS filled with async-to-LDS copies (6 b128 per thread/step).
__global__ __launch_bounds__(256) void gemm_main(const _Float16* __restrict__ X,
                                                 const _Float16* __restrict__ Wt,
                                                 float* __restrict__ out) {
  __shared__ _Float16 lds[2][BUF_HALVES];
  int tid = threadIdx.x;
  int bm = blockIdx.x, bn = blockIdx.y;

  // global->LDS copy assignment: A: 2 threads/row * 32B; B: 1 thread/row * 64B
  int rowA = tid >> 1, segA = tid & 1;
  int rowB = tid;
  const _Float16* gA = X  + (size_t)(bm * BM + rowA) * K_TOT + segA * 16;
  const _Float16* gB = Wt + (size_t)(bn * BN + rowB) * K_TOT;
  uint32_t base0 = (uint32_t)(uintptr_t)&lds[0][0];
  uint32_t base1 = (uint32_t)(uintptr_t)&lds[1][0];
  uint32_t aOff = (uint32_t)(rowA * (LDST * 2) + segA * 32);           // bytes
  uint32_t bOff = (uint32_t)(A_ROWS * (LDST * 2) + rowB * (LDST * 2)); // bytes

  int wave = tid >> 5, lane = tid & 31;
  int wm = wave >> 2, wn = wave & 3;                 // 2 x 4 wave grid
  int ln = lane & 15, g = lane >> 4;

  v8f acc[4][4];
#pragma unroll
  for (int i = 0; i < 4; ++i)
#pragma unroll
    for (int j = 0; j < 4; ++j) acc[i][j] = v8f{};

  const int NITER = K_TOT / 32;                      // 140

#define ISSUE_TILE(bufbase, k)                                   \
  do {                                                           \
    async_b128(gA + (k),      (bufbase) + aOff);                 \
    async_b128(gA + (k) + 8,  (bufbase) + aOff + 16);            \
    async_b128(gB + (k),      (bufbase) + bOff);                 \
    async_b128(gB + (k) + 8,  (bufbase) + bOff + 16);            \
    async_b128(gB + (k) + 16, (bufbase) + bOff + 32);            \
    async_b128(gB + (k) + 24, (bufbase) + bOff + 48);            \
  } while (0)

#define COMPUTE_TILE(Lp)                                                      \
  do {                                                                        \
    const _Float16* L = (Lp);                                                 \
    v16h a[4], b[4];                                                          \
    _Pragma("unroll") for (int i = 0; i < 4; ++i) {                           \
      int m = wm * 64 + i * 16 + ln;                                          \
      *(v8h*)&a[i]       = *(const v8h*)&L[m * LDST + g * 8];                 \
      *((v8h*)&a[i] + 1) = *(const v8h*)&L[m * LDST + 16 + g * 8];            \
    }                                                                         \
    _Pragma("unroll") for (int j = 0; j < 4; ++j) {                           \
      int n = wn * 64 + j * 16 + ln;                                          \
      *(v8h*)&b[j]       = *(const v8h*)&L[A_ROWS * LDST + n * LDST + g * 8];\
      *((v8h*)&b[j] + 1) = *(const v8h*)&L[A_ROWS * LDST + n * LDST + 16 + g * 8];\
    }                                                                         \
    _Pragma("unroll") for (int i = 0; i < 4; ++i)                             \
      _Pragma("unroll") for (int j = 0; j < 4; ++j)                           \
        acc[i][j] = __builtin_amdgcn_wmma_f32_16x16x32_f16(                   \
            false, a[i], false, b[j], (short)0, acc[i][j], false, false);     \
  } while (0)

  ISSUE_TILE(base0, 0);
  int buf = 0;
  for (int i = 0; i < NITER - 1; ++i) {
    uint32_t nbase = (buf == 0) ? base1 : base0;
    ISSUE_TILE(nbase, (i + 1) * 32);                 // prefetch next tile
    wait_asynccnt<6>();                              // current tile landed
    __syncthreads();
    COMPUTE_TILE(&lds[buf][0]);
    __syncthreads();                                 // buf free for next issue
    buf ^= 1;
  }
  wait_asynccnt<0>();
  __syncthreads();
  COMPUTE_TILE(&lds[buf][0]);

#pragma unroll
  for (int i = 0; i < 4; ++i) {
    int row0 = bm * BM + wm * 64 + i * 16 + g * 8;
#pragma unroll
    for (int j = 0; j < 4; ++j) {
      int col = bn * BN + wn * 64 + j * 16 + ln;
#pragma unroll
      for (int r = 0; r < 8; ++r)
        out[(size_t)(row0 + r) * O_OUT + col] = acc[i][j][r];
    }
  }
#undef ISSUE_TILE
#undef COMPUTE_TILE
}

// ------------------------------------------------------------------ launch
extern "C" void kernel_launch(void* const* d_in, const int* in_sizes, int n_in,
                              void* d_out, int out_size, void* d_ws, size_t ws_size,
                              hipStream_t stream) {
  const float* x   = (const float*)d_in[0];
  const float* bW  = (const float*)d_in[1];
  const float* rW  = (const float*)d_in[2];
  const float* A   = (const float*)d_in[3];
  const float* B   = (const float*)d_in[4];
  const float* R   = (const float*)d_in[5];
  float* out = (float*)d_out;
  int N = in_sizes[0] / D_IN;                        // 4096

  _Float16* xcat = (_Float16*)d_ws;                  // [N][4480]
  _Float16* Wt   = xcat + (size_t)N * K_TOT;         // [4096][4480]
  _Float16* Mt   = Wt + (size_t)O_OUT * K_TOT;       // [160][4096]
  float*    S    = (float*)(Mt + (size_t)N_S * D_IN);// [N][160] fp32

  convert_x<<<(N * (D_IN / 4) + 255) / 256, 256, 0, stream>>>(x, xcat, N);

  dim3 g1(O_OUT / 32, D_IN / 32);
  transpose_cvt<<<g1, 256, 0, stream>>>(bW, Wt, O_OUT, K_TOT);
  dim3 g2(O_OUT / 32, N_J / 32);
  transpose_cvt<<<g2, 256, 0, stream>>>(B, Wt + D_IN, O_OUT, K_TOT);

  build_mcat<<<(N_S * D_IN) / 256, 256, 0, stream>>>(rW, R, A, Mt);

  int tiles = (N / 16) * (N_S / 16);
  gemm_prep<<<(tiles + 7) / 8, 256, 0, stream>>>(xcat, Mt, S, N);

  gate_combine<<<(N + 7) / 8, 256, 0, stream>>>(S, xcat, N);

  dim3 g3(N / BM, O_OUT / BN);
  gemm_main<<<g3, 256, 0, stream>>>(xcat, Wt, out);
}
